// Transformer_76751065579543
// MI455X (gfx1250) — compile-verified, matrix operand order checked by reading, not run on hardware
//
#include <hip/hip_runtime.h>
#include <hip/hip_bf16.h>

// Problem constants (match reference)
#define B_  8
#define L_  1024
#define H_  8
#define DK_ 64
#define DV_ 64
#define D_  512
#define DF_ 2048
#define NL_ 6
#define M_  (B_*L_)   // 8192 token rows

typedef __attribute__((ext_vector_type(16))) _Float16 v16h;
typedef __attribute__((ext_vector_type(8)))  float    v8f;
typedef __attribute__((ext_vector_type(2)))  _Float16 h2;

// Guarded CDNA5 async-to-LDS path (ASYNCcnt-tracked DMA staging)
#if defined(__has_builtin)
#if __has_builtin(__builtin_amdgcn_global_load_async_to_lds_b128) && \
    __has_builtin(__builtin_amdgcn_s_wait_asynccnt)
#define USE_ASYNC_LDS 1
#endif
#endif

#ifdef USE_ASYNC_LDS
// Exact pointee type from the toolchain's diagnostic: <4 x i32>
typedef __attribute__((__vector_size__(4 * sizeof(int)))) int v4i_;
typedef __attribute__((address_space(1))) v4i_ GAS1_v4i;   // global
typedef __attribute__((address_space(3))) v4i_ LAS3_v4i;   // LDS
#endif

// K-pair base offset for 16-bit WMMA A/B fragments (ISA 7.12.2):
// VGPR v in lane-half hh holds K = base, base+1 with
// base = (v>=4 ? 16 : 0) + (hh ? 8 : 0) + (v&3)*2
__device__ __forceinline__ int kpair(int v, int hh) {
    return ((v & 4) << 2) + (hh << 3) + ((v & 3) << 1);
}

// ---------------------------------------------------------------------------
// fp32 -> fp16 conversion (weights, once per layer)
// ---------------------------------------------------------------------------
__global__ void cvt_f32_to_f16(const float* __restrict__ s, _Float16* __restrict__ d, int n) {
    int i = blockIdx.x * blockDim.x + threadIdx.x;
    if (i < n) d[i] = (_Float16)s[i];
}

// x -> xbuf (f32 residual stream) and xm16 = (x * mask) in f16 for projections
__global__ void init_x(const float* __restrict__ x, const unsigned char* __restrict__ mask,
                       float* __restrict__ xbuf, _Float16* __restrict__ xm16) {
    int i = blockIdx.x * blockDim.x + threadIdx.x;
    if (i >= M_ * D_) return;
    float val = x[i];
    xbuf[i] = val;
    xm16[i] = mask[i / D_] ? (_Float16)val : (_Float16)0.0f;
}

// ---------------------------------------------------------------------------
// f16 WMMA GEMM:  C[M,N] = scale * (A[M,K] @ B[K,N]) + bias, opt ReLU
// Block = 128 threads (4 wave32). Block tile 128x64; each wave owns 32x64
// (2 A-fragments x 4 B-tiles = 8 WMMAs per K-step, B-fragment reused twice).
// B K-slab double-buffered + transposed in LDS (one barrier per K-step),
// fragment gathers are single b32 ds loads.
// ---------------------------------------------------------------------------
__global__ __launch_bounds__(128) void gemm_wmma_f16(
    const _Float16* __restrict__ A, const _Float16* __restrict__ Bw,
    const float* __restrict__ bias, float* __restrict__ Cf, _Float16* __restrict__ Ch,
    int N, int K, float scale, int relu)
{
    __shared__ _Float16 Bs[2][64][36];   // [buf][n][k] transposed, 72B rows

    const int lane = threadIdx.x & 31;
    const int w    = threadIdx.x >> 5;
    const int hh   = lane >> 4;
    const int ln15 = lane & 15;
    const int nb   = blockIdx.x * 64;
    const int mb   = blockIdx.y * 128 + w * 32;

    const int kr   = threadIdx.x >> 2;        // 0..31: k-row of the slab
    const int nseg = (threadIdx.x & 3) * 16;  // 0,16,32,48: n-segment

    v8f acc[2][4] = {};

    auto stage = [&](int buf, int kb) {
        const _Float16* src = Bw + (size_t)(kb + kr) * N + nb + nseg;
        #pragma unroll
        for (int j = 0; j < 8; ++j) {
            h2 p = *(const h2*)(src + 2 * j);
            Bs[buf][nseg + 2 * j][kr]     = p[0];
            Bs[buf][nseg + 2 * j + 1][kr] = p[1];
        }
    };

    stage(0, 0);
    __syncthreads();

    for (int kb = 0; kb < K; kb += 32) {
        const int buf = (kb >> 5) & 1;
        if (kb + 32 < K) {
            stage(buf ^ 1, kb + 32);          // fill other buffer, no barrier yet
            if (kb + 64 < K)
                __builtin_prefetch(Bw + (size_t)(kb + 64 + kr) * N + nb + nseg, 0, 1);
        }
        // two A fragments (rows mb+0..15, mb+16..31): contiguous b32 pairs
        v16h af[2] = {};
        #pragma unroll
        for (int s = 0; s < 2; ++s) {
            const _Float16* arow = A + (size_t)(mb + s * 16 + ln15) * K + kb;
            #pragma unroll
            for (int v = 0; v < 8; ++v) {
                h2 p = *(const h2*)(arow + kpair(v, hh));
                af[s][2 * v] = p[0]; af[s][2 * v + 1] = p[1];
            }
        }
        // 4 column tiles: one B fragment, two WMMAs (reuse across row sub-tiles)
        #pragma unroll
        for (int t = 0; t < 4; ++t) {
            v16h bf = {};
            const int n = t * 16 + ln15;
            #pragma unroll
            for (int v = 0; v < 8; ++v) {
                h2 p = *(const h2*)(&Bs[buf][n][kpair(v, hh)]);
                bf[2 * v] = p[0]; bf[2 * v + 1] = p[1];
            }
            acc[0][t] = __builtin_amdgcn_wmma_f32_16x16x32_f16(
                false, af[0], false, bf, (short)0, acc[0][t], false, false);
            acc[1][t] = __builtin_amdgcn_wmma_f32_16x16x32_f16(
                false, af[1], false, bf, (short)0, acc[1][t], false, false);
        }
        __syncthreads();   // single barrier per K-step (double buffered)
    }

    // epilogue; C layout: reg r, lane -> row + r + 8*hh, col nb + 16t + ln15
    #pragma unroll
    for (int s = 0; s < 2; ++s) {
        #pragma unroll
        for (int t = 0; t < 4; ++t) {
            const int col = nb + t * 16 + ln15;
            const float bv = bias ? bias[col] : 0.0f;
            #pragma unroll
            for (int r = 0; r < 8; ++r) {
                const int row = mb + s * 16 + r + hh * 8;
                float val = acc[s][t][r] * scale + bv;
                if (relu) val = fmaxf(val, 0.0f);
                if (Cf) Cf[(size_t)row * N + col] = val;
                if (Ch) Ch[(size_t)row * N + col] = (_Float16)val;
            }
        }
    }
}

// ---------------------------------------------------------------------------
// Flash attention: one (b,h) + 64 query rows per block (16 rows per wave).
// Keys streamed in chunks of 32 through LDS (async DMA staging for K when the
// toolchain exposes it); online softmax; key-masked. q16 pre-scaled 1/sqrt(DK).
// ---------------------------------------------------------------------------
__global__ __launch_bounds__(128) void flash_attn(
    const _Float16* __restrict__ Q, const _Float16* __restrict__ Kg,
    const _Float16* __restrict__ Vg, const unsigned char* __restrict__ mask,
    _Float16* __restrict__ Og)
{
    __shared__ _Float16 Ks[32][72];       // [key][dk] row-major, 144B rows (16B aligned)
    __shared__ _Float16 Vt[64][36];       // [dv][key] transposed
    __shared__ _Float16 Ps[4][16][36];    // per-wave P staging (C->A relayout)

    const int lane = threadIdx.x & 31;
    const int w    = threadIdx.x >> 5;
    const int hh   = lane >> 4;
    const int ln15 = lane & 15;
    const int qt   = blockIdx.x & 15;          // L/64 = 16 query tiles
    const int h    = (blockIdx.x >> 4) & 7;
    const int b    = blockIdx.x >> 7;
    const int qbase = qt * 64 + w * 16;

    // Q fragments (two K32 halves of DK=64), wave-private
    v16h qf[2] = {};
    {
        const _Float16* qrow = Q + ((size_t)(b * L_ + qbase + ln15) * H_ + h) * DK_;
        #pragma unroll
        for (int f = 0; f < 2; ++f)
            #pragma unroll
            for (int v = 0; v < 8; ++v) {
                h2 p = *(const h2*)(qrow + f * 32 + kpair(v, hh));
                qf[f][2 * v] = p[0]; qf[f][2 * v + 1] = p[1];
            }
    }

    float mstate[8], lstate[8];
    #pragma unroll
    for (int r = 0; r < 8; ++r) { mstate[r] = -3.0e38f; lstate[r] = 0.0f; }
    v8f accO[4] = {};

    const int ckey = threadIdx.x >> 2;        // 0..31
    const int cseg = (threadIdx.x & 3) * 16;  // 0,16,32,48

    for (int kc = 0; kc < L_; kc += 32) {
        // ---- cooperative stage: K chunk row-major, V chunk transposed ----
        const _Float16* krow = Kg + ((size_t)(b * L_ + kc + ckey) * H_ + h) * DK_ + cseg;
#ifdef USE_ASYNC_LDS
        {
            GAS1_v4i* gp = (GAS1_v4i*)const_cast<_Float16*>(krow);
            LAS3_v4i* lp = (LAS3_v4i*)&Ks[ckey][cseg];
            __builtin_amdgcn_global_load_async_to_lds_b128(gp,     lp,     0, 0);
            __builtin_amdgcn_global_load_async_to_lds_b128(gp + 1, lp + 1, 0, 0);
        }
#else
        {
            #pragma unroll
            for (int j = 0; j < 8; ++j) {
                h2 p = *(const h2*)(krow + 2 * j);
                Ks[ckey][cseg + 2 * j]     = p[0];
                Ks[ckey][cseg + 2 * j + 1] = p[1];
            }
        }
#endif
        {
            const _Float16* vrow = Vg + ((size_t)(b * L_ + kc + ckey) * H_ + h) * DV_ + cseg;
            #pragma unroll
            for (int j = 0; j < 16; ++j) Vt[cseg + j][ckey] = vrow[j];
        }
        if (kc + 32 < L_) {
            __builtin_prefetch(Kg + ((size_t)(b * L_ + kc + 32 + ckey) * H_ + h) * DK_ + cseg, 0, 1);
            __builtin_prefetch(Vg + ((size_t)(b * L_ + kc + 32 + ckey) * H_ + h) * DV_ + cseg, 0, 1);
        }
#ifdef USE_ASYNC_LDS
        __builtin_amdgcn_s_wait_asynccnt(0);
#endif
        __syncthreads();

        // ---- scores: two 16x16 tiles, each = 2 WMMAs over DK=64 ----
        v8f s[2] = {};
        #pragma unroll
        for (int tile = 0; tile < 2; ++tile) {
            const int krow_ = tile * 16 + ln15;
            #pragma unroll
            for (int f = 0; f < 2; ++f) {
                v16h bf = {};
                #pragma unroll
                for (int v = 0; v < 8; ++v) {
                    h2 p = *(const h2*)(&Ks[krow_][f * 32 + kpair(v, hh)]);
                    bf[2 * v] = p[0]; bf[2 * v + 1] = p[1];
                }
                s[tile] = __builtin_amdgcn_wmma_f32_16x16x32_f16(
                    false, qf[f], false, bf, (short)0, s[tile], false, false);
            }
        }
        // key (column) masking
        #pragma unroll
        for (int tile = 0; tile < 2; ++tile) {
            const unsigned char km = mask[b * L_ + kc + tile * 16 + ln15];
            if (!km) {
                #pragma unroll
                for (int r = 0; r < 8; ++r) s[tile][r] = -1.0e30f;
            }
        }
        // ---- online softmax; row = r + 8*hh, reductions in 16-lane halves ----
        #pragma unroll
        for (int r = 0; r < 8; ++r) {
            float mx = fmaxf(s[0][r], s[1][r]);
            #pragma unroll
            for (int off = 1; off < 16; off <<= 1) mx = fmaxf(mx, __shfl_xor(mx, off, 32));
            const float nm = fmaxf(mstate[r], mx);
            const float cf = __expf(mstate[r] - nm);
            float p0 = __expf(s[0][r] - nm);
            float p1 = __expf(s[1][r] - nm);
            s[0][r] = p0; s[1][r] = p1;
            float rs = p0 + p1;
            #pragma unroll
            for (int off = 1; off < 16; off <<= 1) rs += __shfl_xor(rs, off, 32);
            lstate[r] = lstate[r] * cf + rs;
            mstate[r] = nm;
            #pragma unroll
            for (int t = 0; t < 4; ++t) accO[t][r] *= cf;
        }
        // ---- P: C-layout regs -> wave-private LDS -> A-layout fragment ----
        #pragma unroll
        for (int r = 0; r < 8; ++r) {
            const int prow = r + hh * 8;
            Ps[w][prow][ln15]      = (_Float16)s[0][r];
            Ps[w][prow][16 + ln15] = (_Float16)s[1][r];
        }
        v16h pa = {};
        #pragma unroll
        for (int v = 0; v < 8; ++v) {
            h2 p = *(const h2*)(&Ps[w][ln15][kpair(v, hh)]);
            pa[2 * v] = p[0]; pa[2 * v + 1] = p[1];
        }
        // ---- O += P @ V : 4 dv tiles ----
        #pragma unroll
        for (int t = 0; t < 4; ++t) {
            v16h vb = {};
            const int dv = t * 16 + ln15;
            #pragma unroll
            for (int v = 0; v < 8; ++v) {
                h2 p = *(const h2*)(&Vt[dv][kpair(v, hh)]);
                vb[2 * v] = p[0]; vb[2 * v + 1] = p[1];
            }
            accO[t] = __builtin_amdgcn_wmma_f32_16x16x32_f16(
                false, pa, false, vb, (short)0, accO[t], false, false);
        }
        __syncthreads();
    }

    // write O (f16) as [B, L, H*DV]
    #pragma unroll
    for (int t = 0; t < 4; ++t) {
        const int dv = t * 16 + ln15;
        #pragma unroll
        for (int r = 0; r < 8; ++r) {
            const int row = qbase + r + hh * 8;
            const float val = accO[t][r] / lstate[r];
            Og[(size_t)(b * L_ + row) * (H_ * DV_) + h * DV_ + dv] = (_Float16)val;
        }
    }
}

// ---------------------------------------------------------------------------
// res = xin + y * (mask_y ? m : 1); LN(res) -> outf (f32), outh (f16, optionally
// masked), outf2 (optional extra f32, used for final d_out). One row per block.
// ---------------------------------------------------------------------------
__global__ __launch_bounds__(256) void residual_ln(
    const float* __restrict__ xin, const float* __restrict__ y,
    const unsigned char* __restrict__ mask, int mask_y,
    const float* __restrict__ g, const float* __restrict__ bparm,
    float* __restrict__ outf, _Float16* __restrict__ outh, int mask_h,
    float* __restrict__ outf2)
{
    __shared__ float red[256];
    __shared__ float sh_mu, sh_rstd;
    const int row = blockIdx.x;
    const float mrow = mask[row] ? 1.0f : 0.0f;
    const float ymul = mask_y ? mrow : 1.0f;
    const size_t base = (size_t)row * D_;
    const int i0 = threadIdx.x, i1 = threadIdx.x + 256;

    float v0 = xin[base + i0] + y[base + i0] * ymul;
    float v1 = xin[base + i1] + y[base + i1] * ymul;

    red[threadIdx.x] = v0 + v1;
    __syncthreads();
    for (int s = 128; s > 0; s >>= 1) {
        if (threadIdx.x < s) red[threadIdx.x] += red[threadIdx.x + s];
        __syncthreads();
    }
    if (threadIdx.x == 0) sh_mu = red[0] / (float)D_;
    __syncthreads();
    const float mu = sh_mu;
    const float d0 = v0 - mu, d1 = v1 - mu;

    red[threadIdx.x] = d0 * d0 + d1 * d1;
    __syncthreads();
    for (int s = 128; s > 0; s >>= 1) {
        if (threadIdx.x < s) red[threadIdx.x] += red[threadIdx.x + s];
        __syncthreads();
    }
    if (threadIdx.x == 0) sh_rstd = rsqrtf(red[0] / (float)D_ + 1e-5f);
    __syncthreads();
    const float rstd = sh_rstd;
    const float hmul = mask_h ? mrow : 1.0f;

    const float o0 = d0 * rstd * g[i0] + bparm[i0];
    const float o1 = d1 * rstd * g[i1] + bparm[i1];
    outf[base + i0] = o0; outf[base + i1] = o1;
    outh[base + i0] = (_Float16)(o0 * hmul);
    outh[base + i1] = (_Float16)(o1 * hmul);
    if (outf2) { outf2[base + i0] = o0; outf2[base + i1] = o1; }
}

// ---------------------------------------------------------------------------
extern "C" void kernel_launch(void* const* d_in, const int* in_sizes, int n_in,
                              void* d_out, int out_size, void* d_ws, size_t ws_size,
                              hipStream_t stream) {
    (void)in_sizes; (void)n_in; (void)out_size; (void)ws_size;
    const float* x    = (const float*)d_in[0];
    const float* Wq   = (const float*)d_in[1];
    const float* Wk   = (const float*)d_in[2];
    const float* Wv   = (const float*)d_in[3];
    const float* Wo   = (const float*)d_in[4];
    const float* ln1g = (const float*)d_in[5];
    const float* ln1b = (const float*)d_in[6];
    const float* W1   = (const float*)d_in[7];
    const float* b1   = (const float*)d_in[8];
    const float* W2   = (const float*)d_in[9];
    const float* b2   = (const float*)d_in[10];
    const float* ln2g = (const float*)d_in[11];
    const float* ln2b = (const float*)d_in[12];
    const unsigned char* mask = (const unsigned char*)d_in[13];

    // carve workspace (total ~140 MB)
    char* p = (char*)d_ws;
    auto carve = [&](size_t bytes) -> void* {
        void* r = (void*)p; p += (bytes + 255) & ~(size_t)255; return r;
    };
    const size_t MD = (size_t)M_ * D_;
    float*    xbuf  = (float*)carve(MD * 4);
    float*    x1buf = (float*)carve(MD * 4);
    float*    ybuf  = (float*)carve(MD * 4);
    _Float16* xm16  = (_Float16*)carve(MD * 2);
    _Float16* x116  = (_Float16*)carve(MD * 2);
    _Float16* qh    = (_Float16*)carve(MD * 2);
    _Float16* kh    = (_Float16*)carve(MD * 2);
    _Float16* vh    = (_Float16*)carve(MD * 2);
    _Float16* oh    = (_Float16*)carve(MD * 2);
    _Float16* hbuf  = (_Float16*)carve((size_t)M_ * DF_ * 2);
    _Float16* wq16  = (_Float16*)carve((size_t)D_ * D_ * 2);
    _Float16* wk16  = (_Float16*)carve((size_t)D_ * D_ * 2);
    _Float16* wv16  = (_Float16*)carve((size_t)D_ * D_ * 2);
    _Float16* wo16  = (_Float16*)carve((size_t)D_ * D_ * 2);
    _Float16* w1h   = (_Float16*)carve((size_t)D_ * DF_ * 2);
    _Float16* w2h   = (_Float16*)carve((size_t)D_ * DF_ * 2);

    init_x<<<(int)((MD + 255) / 256), 256, 0, stream>>>(x, mask, xbuf, xm16);

    const float qscale = 0.125f;   // 1/sqrt(DK)
    const int nDD = D_ * D_, nDF = D_ * DF_;

    for (int i = 0; i < NL_; ++i) {
        const size_t wAoff = (size_t)i * D_ * D_;   // H*DK == D
        const size_t wFoff = (size_t)i * D_ * DF_;
        cvt_f32_to_f16<<<(nDD + 255) / 256, 256, 0, stream>>>(Wq + wAoff, wq16, nDD);
        cvt_f32_to_f16<<<(nDD + 255) / 256, 256, 0, stream>>>(Wk + wAoff, wk16, nDD);
        cvt_f32_to_f16<<<(nDD + 255) / 256, 256, 0, stream>>>(Wv + wAoff, wv16, nDD);
        cvt_f32_to_f16<<<(nDD + 255) / 256, 256, 0, stream>>>(Wo + wAoff, wo16, nDD);
        cvt_f32_to_f16<<<(nDF + 255) / 256, 256, 0, stream>>>(W1 + wFoff, w1h, nDF);
        cvt_f32_to_f16<<<(nDF + 255) / 256, 256, 0, stream>>>(W2 + wFoff, w2h, nDF);

        dim3 gA(D_ / 64, M_ / 128);          // (8, 64)
        gemm_wmma_f16<<<gA, 128, 0, stream>>>(xm16, wq16, nullptr, nullptr, qh, D_, D_, qscale, 0);
        gemm_wmma_f16<<<gA, 128, 0, stream>>>(xm16, wk16, nullptr, nullptr, kh, D_, D_, 1.0f, 0);
        gemm_wmma_f16<<<gA, 128, 0, stream>>>(xm16, wv16, nullptr, nullptr, vh, D_, D_, 1.0f, 0);

        flash_attn<<<B_ * H_ * (L_ / 64), 128, 0, stream>>>(qh, kh, vh, mask, oh);

        gemm_wmma_f16<<<gA, 128, 0, stream>>>(oh, wo16, nullptr, ybuf, nullptr, D_, D_, 1.0f, 0);
        residual_ln<<<M_, 256, 0, stream>>>(xbuf, ybuf, mask, 1,
                                            ln1g + (size_t)i * D_, ln1b + (size_t)i * D_,
                                            x1buf, x116, 0, nullptr);

        dim3 gF1(DF_ / 64, M_ / 128);        // (32, 64)
        gemm_wmma_f16<<<gF1, 128, 0, stream>>>(x116, w1h, b1 + (size_t)i * DF_,
                                               nullptr, hbuf, DF_, D_, 1.0f, 1);
        gemm_wmma_f16<<<gA, 128, 0, stream>>>(hbuf, w2h, b2 + (size_t)i * D_,
                                              ybuf, nullptr, D_, DF_, 1.0f, 0);
        residual_ln<<<M_, 256, 0, stream>>>(x1buf, ybuf, mask, 0,
                                            ln2g + (size_t)i * D_, ln2b + (size_t)i * D_,
                                            xbuf, xm16, 1,
                                            (i == NL_ - 1) ? (float*)d_out : nullptr);
    }
}
